// BiDirectionalMinGRU_20452634263722
// MI455X (gfx1250) — matrix operand order repaired and checked by reading, not compiled, not run on hardware
//
#include <hip/hip_runtime.h>
#include <hip/hip_fp16.h>

typedef __attribute__((ext_vector_type(16))) _Float16 v16h;
typedef __attribute__((ext_vector_type(8)))  _Float16 v8h;
typedef __attribute__((ext_vector_type(8)))  float    v8f;

static constexpr int kB   = 32;
static constexpr int kL   = 2048;
static constexpr int kM   = kB * kL;   // 65536 rows for all GEMMs
static constexpr int kH   = 256;       // hidden
static constexpr int kTE  = 64;        // time-enc dims
static constexpr int kKXC = 96;        // 66 padded to 96 (3 x K32 chunks)
static constexpr int kGH  = 128;       // gauss head hidden
static constexpr int kBI  = 576;       // 2H + TE, already multiple of 32

// ---------------------------------------------------------------------------
// Weight convert: Wt[n*KPAD + k] = (k<K) ? W[k*N + n] : 0   (fp32 -> fp16, T)
// ---------------------------------------------------------------------------
__global__ void convert_wt_kernel(const float* __restrict__ W,
                                  _Float16* __restrict__ Wt,
                                  int K, int N, int KPAD) {
  int gid = blockIdx.x * blockDim.x + threadIdx.x;
  if (gid >= N * KPAD) return;
  int n = gid / KPAD, k = gid % KPAD;
  float v = (k < K) ? W[(size_t)k * N + n] : 0.0f;
  Wt[(size_t)n * KPAD + k] = (_Float16)v;
}

// ---------------------------------------------------------------------------
// Prep: A1[pos][j] = relu(t[pos]*te_w1[j] + te_b1[j])  (fp16, 65536 x 64)
// Also fill xc cols 0..1 from x and zero the K-pad cols 66..95.
// One thread per (pos, j), j in [0,64).
// ---------------------------------------------------------------------------
__global__ void prep_kernel(const float* __restrict__ x,
                            const float* __restrict__ t,
                            const float* __restrict__ w1,
                            const float* __restrict__ b1,
                            _Float16* __restrict__ A1,
                            _Float16* __restrict__ xc) {
  int gid = blockIdx.x * blockDim.x + threadIdx.x;   // kM*64 threads exactly
  int pos = gid >> 6;
  int j   = gid & 63;
  float tv = t[pos];
  float v  = fmaxf(tv * w1[j] + b1[j], 0.0f);
  A1[(size_t)pos * kTE + j] = (_Float16)v;
  if (j < 2)   xc[(size_t)pos * kKXC + j]      = (_Float16)x[(size_t)pos * 2 + j];
  if (j >= 34) xc[(size_t)pos * kKXC + 32 + j] = (_Float16)0.0f;  // cols 66..95
}

// ---------------------------------------------------------------------------
// Epilogue: bias + activation + fp16 store of one 16x16 C tile.
// C layout (f32 16x16): VGPR r -> row = (lane>=16 ? 8 : 0) + r, col = lane&15.
// ---------------------------------------------------------------------------
template <int ACT>
__device__ __forceinline__ void
store_tile(const v8f& c, int rowBase, int colBase, int hs, int l15,
           const float* __restrict__ bias,
           _Float16* __restrict__ o16a, int lda, int offa,
           _Float16* __restrict__ o16b, int ldb, int offb) {
  const int col = colBase + l15;
  const float bv = bias[col];
#pragma unroll
  for (int r = 0; r < 8; ++r) {
    float v = c[r] + bv;
    if (ACT == 1)      v = fmaxf(v, 0.0f);
    else if (ACT == 2) v = 1.0f / (1.0f + __expf(-v));
    else if (ACT == 3) v = tanhf(v);
    const size_t row = (size_t)(rowBase + hs * 8 + r);
    const _Float16 hv = (_Float16)v;
    o16a[row * lda + offa + col] = hv;
    if (o16b) o16b[row * ldb + offb + col] = hv;
  }
}

// ---------------------------------------------------------------------------
// WMMA GEMM: out = act(A(M x K) @ Wt^T + bias), fp16 in, f32 accum, fp16 out.
// A row-major, row stride K (padded, multiple of 32). Wt transposed: Wt[n*K+k].
// Block = 128 threads = 4 waves. Each wave owns a 32x32 C macro-tile (2x2
// register tiles, 4 accumulators) -> A/B fragments each reused twice:
// 8 b128 loads feed 4 WMMAs per K32 step (2 loads/WMMA vs 4 for 1x1 tiling).
// Waves arranged 2x2 -> block covers 64x64 of C.  Grid = (M/64, N/64).
// ACT: 0 none, 1 relu, 2 sigmoid, 3 tanh.
// Optional second fp16 destination (scatters t_enc into xc AND h_bi).
// ---------------------------------------------------------------------------
template <int ACT>
__global__ void __launch_bounds__(128)
gemm_wmma_kernel(const _Float16* __restrict__ A,
                 const _Float16* __restrict__ Wt,
                 const float* __restrict__ bias, int K,
                 _Float16* __restrict__ o16a, int lda, int offa,
                 _Float16* __restrict__ o16b, int ldb, int offb) {
  const int lane = threadIdx.x & 31;
  const int wave = threadIdx.x >> 5;
  const int hs   = lane >> 4;     // half-wave select
  const int l15  = lane & 15;
  const int row0 = blockIdx.x * 64 + (wave >> 1) * 32;
  const int col0 = blockIdx.y * 64 + (wave & 1) * 32;

  v8f c00 = {}, c01 = {}, c10 = {}, c11 = {};
  // A fragment (16x32 f16): lane<16 row=l15, halves0-7=K0-7, halves8-15=K16-23;
  // lanes 16-31 shift K by +8.  Two contiguous b128 loads per fragment.
  const _Float16* aRow0 = A + (size_t)(row0 + l15) * K + hs * 8;
  const _Float16* aRow1 = aRow0 + (size_t)16 * K;
  // B fragment (32x16): lane<16 N=l15 K0-7,K16-23; lanes 16-31 shift K by +8 (Wt is NxK).
  const _Float16* bRow0 = Wt + (size_t)(col0 + l15) * K + hs * 8;
  const _Float16* bRow1 = bRow0 + (size_t)16 * K;

  for (int k0 = 0; k0 < K; k0 += 32) {
    union { v16h v; v8h h[2]; } a0, a1, b0, b1;
    a0.h[0] = *(const v8h*)(aRow0 + k0);
    a0.h[1] = *(const v8h*)(aRow0 + k0 + 16);
    a1.h[0] = *(const v8h*)(aRow1 + k0);
    a1.h[1] = *(const v8h*)(aRow1 + k0 + 16);
    b0.h[0] = *(const v8h*)(bRow0 + k0);
    b0.h[1] = *(const v8h*)(bRow0 + k0 + 16);
    b1.h[0] = *(const v8h*)(bRow1 + k0);
    b1.h[1] = *(const v8h*)(bRow1 + k0 + 16);
    c00 = __builtin_amdgcn_wmma_f32_16x16x32_f16(false, a0.v, false, b0.v,
                                                 (short)0, c00, false, false);
    c01 = __builtin_amdgcn_wmma_f32_16x16x32_f16(false, a0.v, false, b1.v,
                                                 (short)0, c01, false, false);
    c10 = __builtin_amdgcn_wmma_f32_16x16x32_f16(false, a1.v, false, b0.v,
                                                 (short)0, c10, false, false);
    c11 = __builtin_amdgcn_wmma_f32_16x16x32_f16(false, a1.v, false, b1.v,
                                                 (short)0, c11, false, false);
  }

  store_tile<ACT>(c00, row0,      col0,      hs, l15, bias, o16a, lda, offa, o16b, ldb, offb);
  store_tile<ACT>(c01, row0,      col0 + 16, hs, l15, bias, o16a, lda, offa, o16b, ldb, offb);
  store_tile<ACT>(c10, row0 + 16, col0,      hs, l15, bias, o16a, lda, offa, o16b, ldb, offb);
  store_tile<ACT>(c11, row0 + 16, col0 + 16, hs, l15, bias, o16a, lda, offa, o16b, ldb, offb);
}

// ---------------------------------------------------------------------------
// MinGRU linear scan: one thread per (b, h) recurrence (8192 total).
// Stores h_prev (state BEFORE update) at each step, matching the reference
// (lax.scan returns the pre-update carry at each index, fwd and reverse).
// Writes fp16 directly into h_bi at column offset coloff.
// ---------------------------------------------------------------------------
__global__ void scan_kernel(const _Float16* __restrict__ z,
                            const _Float16* __restrict__ ht,
                            _Float16* __restrict__ hbi,
                            int coloff, int reverse) {
  int idx = blockIdx.x * blockDim.x + threadIdx.x;   // kB*kH threads exactly
  int b = idx / kH, h = idx % kH;
  float hc = 0.0f;
  if (!reverse) {
    for (int t = 0; t < kL; ++t) {
      size_t p = ((size_t)b * kL + t) * kH + h;
      hbi[((size_t)b * kL + t) * kBI + coloff + h] = (_Float16)hc;
      float zt = (float)z[p], htt = (float)ht[p];
      hc = (1.0f - zt) * hc + zt * htt;
    }
  } else {
    for (int t = kL - 1; t >= 0; --t) {
      size_t p = ((size_t)b * kL + t) * kH + h;
      hbi[((size_t)b * kL + t) * kBI + coloff + h] = (_Float16)hc;
      float zt = (float)z[p], htt = (float)ht[p];
      hc = (1.0f - zt) * hc + zt * htt;
    }
  }
}

// ---------------------------------------------------------------------------
// Final head: out[row] = dot(g[row, 0:128], gh_w2) + gh_b2.  One wave / row.
// ---------------------------------------------------------------------------
__global__ void final_dot_kernel(const _Float16* __restrict__ g,
                                 const float* __restrict__ w2,
                                 const float* __restrict__ b2,
                                 float* __restrict__ out) {
  int row  = (blockIdx.x * blockDim.x + threadIdx.x) >> 5;
  int lane = threadIdx.x & 31;
  const _Float16* gp = g + (size_t)row * kGH + lane * 4;
  float acc = 0.0f;
#pragma unroll
  for (int i = 0; i < 4; ++i) acc += (float)gp[i] * w2[lane * 4 + i];
#pragma unroll
  for (int m = 16; m >= 1; m >>= 1) acc += __shfl_xor(acc, m, 32);
  if (lane == 0) out[row] = acc + b2[0];
}

// ---------------------------------------------------------------------------
extern "C" void kernel_launch(void* const* d_in, const int* in_sizes, int n_in,
                              void* d_out, int out_size, void* d_ws, size_t ws_size,
                              hipStream_t stream) {
  (void)in_sizes; (void)n_in; (void)out_size; (void)ws_size;

  const float* x       = (const float*)d_in[0];
  const float* t       = (const float*)d_in[1];
  const float* te_w1   = (const float*)d_in[2];
  const float* te_b1   = (const float*)d_in[3];
  const float* te_w2   = (const float*)d_in[4];
  const float* te_b2   = (const float*)d_in[5];
  const float* fproj_w = (const float*)d_in[6];
  const float* fproj_b = (const float*)d_in[7];
  const float* bproj_w = (const float*)d_in[8];
  const float* bproj_b = (const float*)d_in[9];
  const float* fwz_w   = (const float*)d_in[10];
  const float* fwz_b   = (const float*)d_in[11];
  const float* fwh_w   = (const float*)d_in[12];
  const float* fwh_b   = (const float*)d_in[13];
  const float* bwz_w   = (const float*)d_in[14];
  const float* bwz_b   = (const float*)d_in[15];
  const float* bwh_w   = (const float*)d_in[16];
  const float* bwh_b   = (const float*)d_in[17];
  const float* gh_w1   = (const float*)d_in[18];
  const float* gh_b1   = (const float*)d_in[19];
  const float* gh_w2   = (const float*)d_in[20];
  const float* gh_b2   = (const float*)d_in[21];
  float* out = (float*)d_out;

  // ---- carve workspace (all chunks 256B aligned) ----
  char* wp = (char*)d_ws;
  auto carve = [&](size_t bytes) {
    void* p = (void*)wp;
    wp += (bytes + 255) & ~(size_t)255;
    return p;
  };
  _Float16* te_w2t = (_Float16*)carve((size_t)kTE * kTE * 2);
  _Float16* fprojt = (_Float16*)carve((size_t)kH * kKXC * 2);
  _Float16* bprojt = (_Float16*)carve((size_t)kH * kKXC * 2);
  _Float16* fwzt   = (_Float16*)carve((size_t)kH * kH * 2);
  _Float16* fwht   = (_Float16*)carve((size_t)kH * kH * 2);
  _Float16* bwzt   = (_Float16*)carve((size_t)kH * kH * 2);
  _Float16* bwht   = (_Float16*)carve((size_t)kH * kH * 2);
  _Float16* ghw1t  = (_Float16*)carve((size_t)kGH * kBI * 2);
  _Float16* A1     = (_Float16*)carve((size_t)kM * kTE * 2);
  _Float16* xc     = (_Float16*)carve((size_t)kM * kKXC * 2);
  _Float16* hbi    = (_Float16*)carve((size_t)kM * kBI * 2);
  _Float16* inp_f  = (_Float16*)carve((size_t)kM * kH * 2);
  _Float16* inp_b  = (_Float16*)carve((size_t)kM * kH * 2);
  _Float16* zf     = (_Float16*)carve((size_t)kM * kH * 2);
  _Float16* htf    = (_Float16*)carve((size_t)kM * kH * 2);
  _Float16* zb     = (_Float16*)carve((size_t)kM * kH * 2);
  _Float16* htb    = (_Float16*)carve((size_t)kM * kH * 2);
  _Float16* g      = (_Float16*)carve((size_t)kM * kGH * 2);

  // ---- 1) convert / transpose / pad weights to fp16 ----
  auto cvt = [&](const float* W, _Float16* Wt, int K, int N, int KPAD) {
    int n = N * KPAD;
    convert_wt_kernel<<<(n + 255) / 256, 256, 0, stream>>>(W, Wt, K, N, KPAD);
  };
  cvt(te_w2,   te_w2t, kTE, kTE, kTE);
  cvt(fproj_w, fprojt, 66,  kH,  kKXC);
  cvt(bproj_w, bprojt, 66,  kH,  kKXC);
  cvt(fwz_w,   fwzt,   kH,  kH,  kH);
  cvt(fwh_w,   fwht,   kH,  kH,  kH);
  cvt(bwz_w,   bwzt,   kH,  kH,  kH);
  cvt(bwh_w,   bwht,   kH,  kH,  kH);
  cvt(gh_w1,   ghw1t,  kBI, kGH, kBI);

  // ---- 2) time-enc layer1 + xc fill ----
  prep_kernel<<<(kM * 64) / 256, 256, 0, stream>>>(x, t, te_w1, te_b1, A1, xc);

  // ---- 3) t_enc = A1 @ te_w2 + te_b2 -> xc cols 2..65 AND hbi cols 512..575
  gemm_wmma_kernel<0><<<dim3(kM / 64, kTE / 64), 128, 0, stream>>>(
      A1, te_w2t, te_b2, kTE, xc, kKXC, 2, hbi, kBI, 512);

  // ---- 4) input projections (K = 96 padded) ----
  gemm_wmma_kernel<0><<<dim3(kM / 64, kH / 64), 128, 0, stream>>>(
      xc, fprojt, fproj_b, kKXC, inp_f, kH, 0, (_Float16*)nullptr, 0, 0);
  gemm_wmma_kernel<0><<<dim3(kM / 64, kH / 64), 128, 0, stream>>>(
      xc, bprojt, bproj_b, kKXC, inp_b, kH, 0, (_Float16*)nullptr, 0, 0);

  // ---- 5) gates: z = sigmoid(...), h_tilde = tanh(...) ----
  gemm_wmma_kernel<2><<<dim3(kM / 64, kH / 64), 128, 0, stream>>>(
      inp_f, fwzt, fwz_b, kH, zf, kH, 0, (_Float16*)nullptr, 0, 0);
  gemm_wmma_kernel<3><<<dim3(kM / 64, kH / 64), 128, 0, stream>>>(
      inp_f, fwht, fwh_b, kH, htf, kH, 0, (_Float16*)nullptr, 0, 0);
  gemm_wmma_kernel<2><<<dim3(kM / 64, kH / 64), 128, 0, stream>>>(
      inp_b, bwzt, bwz_b, kH, zb, kH, 0, (_Float16*)nullptr, 0, 0);
  gemm_wmma_kernel<3><<<dim3(kM / 64, kH / 64), 128, 0, stream>>>(
      inp_b, bwht, bwh_b, kH, htb, kH, 0, (_Float16*)nullptr, 0, 0);

  // ---- 6) sequential scans -> hbi cols 0..255 (fwd) and 256..511 (bwd) ----
  scan_kernel<<<(kB * kH) / 256, 256, 0, stream>>>(zf, htf, hbi, 0,  0);
  scan_kernel<<<(kB * kH) / 256, 256, 0, stream>>>(zb, htb, hbi, kH, 1);

  // ---- 7) head hidden: g = relu(hbi @ gh_w1 + gh_b1) ----
  gemm_wmma_kernel<1><<<dim3(kM / 64, kGH / 64), 128, 0, stream>>>(
      hbi, ghw1t, gh_b1, kBI, g, kGH, 0, (_Float16*)nullptr, 0, 0);

  // ---- 8) final dot ----
  final_dot_kernel<<<kM / 8, 256, 0, stream>>>(g, gh_w2, gh_b2, out);
}
